// JcStrucEncoder_30185030156490
// MI455X (gfx1250) — matrix operand order repaired and checked by reading, not compiled, not run on hardware
//
#include <hip/hip_runtime.h>
#include <hip/hip_bf16.h>
#include <math.h>

// ---------------------------------------------------------------------------
// Fused per-sequence transformer + Sinkhorn-OT forward for gfx1250 (MI455X).
// One workgroup (128 threads = 4 wave32) per sequence; all state in LDS.
// GEMMs use v_wmma_f32_16x16x32_f16 with B pre-swizzled into the per-lane
// WMMA register layout (contiguous 16B LDS loads for both operands).
// Attention (DHEAD=2) uses fused online softmax on VALU; Sinkhorn's 100
// iterations run in-LDS with a 4-way split logsumexp reduction.
// ---------------------------------------------------------------------------

#define BLOCK   128
#define NWAVES  4
#define NTOK    100
#define NPAD    112
#define EMBD    64
#define HEADS   3
#define INNER   6
#define QKVN    18
#define MLPD    16
#define DEPTHL  4
#define OUTD    32
#define OTM     25
#define OTI     100

typedef __attribute__((ext_vector_type(16))) _Float16 v16h;
typedef __attribute__((ext_vector_type(8)))  _Float16 v8h;
typedef __attribute__((ext_vector_type(8)))  float    v8f;

// --- WMMA operand loaders (ISA 7.12.2 layouts, wave32) ----------------------
// A tile 16x32 f16: lanes 0-15 row M=lane hold K={0..7,16..23}; lanes 16-31
// hold K={8..15,24..31}.  Two contiguous 16B LDS loads per lane.
__device__ __forceinline__ v16h wmma_load_a(const _Float16* A, int lda) {
  int lane = threadIdx.x & 31;
  int r  = lane & 15;
  int kb = (lane >> 4) << 3;               // 0 or 8
  const _Float16* p = A + r * lda + kb;
  v8h lo = *(const v8h*)p;
  v8h hi = *(const v8h*)(p + 16);
  v16h a;
#pragma unroll
  for (int i = 0; i < 8; ++i) { a[i] = lo[i]; a[i + 8] = hi[i]; }
  return a;
}

// B operand is staged pre-swizzled: 16 contiguous halfs per lane per
// (kstep, ntile): dst[((kk*nt + ntile)*32 + lane)*16 + i] holds
// B[k = kk*32 + (lane>=16 ? 8:0) + (i&7) + (i>=8 ? 16:0)][n = ntile*16 + (lane&15)].
__device__ __forceinline__ v16h wmma_load_b_swz(const _Float16* Bs, int nt,
                                                int kk, int ntile) {
  int lane = threadIdx.x & 31;
  const _Float16* p = Bs + (((kk * nt + ntile) * 32 + lane) << 4);
  v8h lo = *(const v8h*)p;
  v8h hi = *(const v8h*)(p + 8);
  v16h b;
#pragma unroll
  for (int i = 0; i < 8; ++i) { b[i] = lo[i]; b[i + 8] = hi[i]; }
  return b;
}

// C/D tile 16x16 f32: lanes 0-15 N=lane, VGPR i -> M=i; lanes 16-31 -> M=8+i.
__device__ __forceinline__ void wmma_store_c(float* C, int ldc, v8f acc) {
  int lane = threadIdx.x & 31;
  int nn = lane & 15;
  int mb = (lane >> 4) << 3;
#pragma unroll
  for (int i = 0; i < 8; ++i) C[(mb + i) * ldc + nn] = acc[i];
}

// Stage B (f32 source, global or LDS) into the swizzled f16 layout above,
// with zero padding outside [Kreal, Nreal).  transp=true reads src[n][k].
__device__ void stage_b_swz(_Float16* dst, const float* src, int srcLd,
                            int K, int N, int Kreal, int Nreal, bool transp) {
  int nt = N >> 4;
  int total = (K >> 5) * nt * 512;         // kt * nt * 32 lanes * 16 halfs
  for (int idx = threadIdx.x; idx < total; idx += BLOCK) {
    int i    = idx & 15;
    int lane = (idx >> 4) & 31;
    int t    = idx >> 9;
    int ntile = t % nt;
    int kk    = t / nt;
    int k = (kk << 5) + ((lane & 16) >> 1) + (i & 7) + ((i & 8) << 1);   // (lane>=16 ? 8 : 0) + (i<8 ? i : 16+i-8)
    int n = (ntile << 4) + (lane & 15);
    float v = 0.f;
    if (k < Kreal && n < Nreal)
      v = transp ? src[n * srcLd + k] : src[k * srcLd + n];
    dst[idx] = (_Float16)v;
  }
}

// LDS GEMM: C[MxN] = A[MxK] * Bswz, f16 in / f32 out. M%16==0, N%16==0,
// K%32==0. Tiles distributed over the block's 4 waves (uniform control flow,
// EXEC stays all-ones as WMMA requires).
__device__ void lds_gemm(const _Float16* A, int lda, const _Float16* Bs,
                         float* C, int ldc, int M, int N, int K) {
  int wave = threadIdx.x >> 5;
  int mt = M >> 4, nt = N >> 4, kt = K >> 5;
  for (int t = wave; t < mt * nt; t += NWAVES) {
    int m0    = (t % mt) << 4;
    int ntile = t / mt;
    v8f acc = {};
    for (int kk = 0; kk < kt; ++kk) {
      v16h a = wmma_load_a(A + m0 * lda + (kk << 5), lda);
      v16h b = wmma_load_b_swz(Bs, nt, kk, ntile);
      acc = __builtin_amdgcn_wmma_f32_16x16x32_f16(
          false, a, false, b, (short)0, acc, false, false);
    }
    wmma_store_c(C + m0 * ldc + (ntile << 4), ldc, acc);
  }
}

struct Smem {
  alignas(16) float     xs[NPAD * EMBD];     // residual stream (fp32)
  alignas(16) float     tmp[NPAD * EMBD];    // gemm outputs / xd / Kmat
  alignas(16) _Float16  ah[NPAD * EMBD];     // A operand staging (f16)
  alignas(16) _Float16  bh[64 * 64];         // B operand staging (swizzled)
  float obuf[NPAD * 8];                      // attention output (6 used)
  float u[NPAD];
  float vv[32];
  float loga[NPAD];
  float pmx[4 * 32];                         // Sinkhorn partial max
  float psum[4 * 32];                        // Sinkhorn partial sum
  int   msk[NPAD];
  float cnt;
};

__device__ void layernorm_to_f16(const float* x, _Float16* dst,
                                 const float* sc, const float* bi) {
  for (int r = threadIdx.x; r < NPAD; r += BLOCK) {
    if (r < NTOK) {
      const float* row = x + r * EMBD;
      float mu = 0.f;
      for (int c = 0; c < EMBD; ++c) mu += row[c];
      mu *= (1.f / EMBD);
      float va = 0.f;
      for (int c = 0; c < EMBD; ++c) { float d = row[c] - mu; va += d * d; }
      va *= (1.f / EMBD);
      float rs = rsqrtf(va + 1e-5f);
      for (int c = 0; c < EMBD; ++c)
        dst[r * EMBD + c] = (_Float16)((row[c] - mu) * rs * sc[c] + bi[c]);
    } else {
      for (int c = 0; c < EMBD; ++c) dst[r * EMBD + c] = (_Float16)0.f;
    }
  }
}

__global__ __launch_bounds__(BLOCK)
void seq_forward(const float* __restrict__ jc,    const int*   __restrict__ flg,
                 const float* __restrict__ enc_w, const float* __restrict__ enc_b,
                 const float* __restrict__ B_sig, const float* __restrict__ B_jid,
                 const float* __restrict__ qkv_w, const float* __restrict__ out_w,
                 const float* __restrict__ out_b, const float* __restrict__ ln1_s,
                 const float* __restrict__ ln1_b, const float* __restrict__ ln2_s,
                 const float* __restrict__ ln2_b, const float* __restrict__ ff1_w,
                 const float* __restrict__ ff1_b, const float* __restrict__ ff2_w,
                 const float* __restrict__ ff2_b, const float* __restrict__ dec_w,
                 const float* __restrict__ dec_b, const float* __restrict__ ot_w,
                 float* __restrict__ out) {
  __shared__ Smem sm;
  const int s   = blockIdx.x;
  const int tid = threadIdx.x;
  const float TWO_PI = 6.283185307179586f;

  // ---- mask / log_a --------------------------------------------------------
  for (int r = tid; r < NPAD; r += BLOCK)
    sm.msk[r] = (r < NTOK) ? (flg[s * NTOK + r] > 0) : 0;
  __syncthreads();
  if (tid == 0) {
    int c = 0;
    for (int r = 0; r < NTOK; ++r) c += sm.msk[r];
    sm.cnt = (float)c;
  }
  __syncthreads();
  for (int r = tid; r < NPAD; r += BLOCK)
    sm.loga[r] = (r < NTOK && sm.msk[r]) ? -logf(sm.cnt) : -1.0e9f;

  // ---- embedding: enc + fourier(sig) + fourier(joint id) -------------------
  for (int idx = tid; idx < NPAD * EMBD; idx += BLOCK) {
    int r = idx / EMBD, c = idx % EMBD;
    float val = 0.f;
    if (r < NTOK) {
      const float x0 = jc[(s * NTOK + r) * 3 + 0];
      const float x1 = jc[(s * NTOK + r) * 3 + 1];
      const float x2 = jc[(s * NTOK + r) * 3 + 2];
      val = x0 * enc_w[c] + x1 * enc_w[EMBD + c] + x2 * enc_w[2 * EMBD + c] + enc_b[c];
      int cc = c & 31;
      float pr = TWO_PI * (x0 * B_sig[cc] + x1 * B_sig[32 + cc] + x2 * B_sig[64 + cc]);
      val += (c < 32) ? sinf(pr) : cosf(pr);
      float pj = TWO_PI * (float)(r >> 2) * B_jid[cc];
      val += (c < 32) ? sinf(pj) : cosf(pj);
    }
    sm.xs[idx] = val;
  }
  __syncthreads();

  // ---- transformer layers --------------------------------------------------
  for (int l = 0; l < DEPTHL; ++l) {
    // h = LN1(x) -> ah (f16, zero-padded rows)
    layernorm_to_f16(sm.xs, sm.ah, ln1_s + l * EMBD, ln1_b + l * EMBD);
    // qkv_w[l] (64x18 -> 64x32 padded, swizzled)
    stage_b_swz(sm.bh, qkv_w + l * EMBD * QKVN, QKVN, EMBD, 32, EMBD, QKVN, false);
    __syncthreads();
    lds_gemm(sm.ah, EMBD, sm.bh, sm.tmp, 32, NPAD, 32, EMBD);       // qkv
    __syncthreads();

    // fused attention, DHEAD=2 (online softmax; kmask on keys)
    const float scale = 0.7071067811865476f;
    for (int t = tid; t < HEADS * NTOK; t += BLOCK) {
      int h = t / NTOK, i = t % NTOK;
      float q0 = sm.tmp[i * 32 + h * 2 + 0];
      float q1 = sm.tmp[i * 32 + h * 2 + 1];
      float mx = -3.0e38f, ssum = 0.f, o0 = 0.f, o1 = 0.f;
      for (int j = 0; j < NTOK; ++j) {
        float k0 = sm.tmp[j * 32 + 6 + h * 2 + 0];
        float k1 = sm.tmp[j * 32 + 6 + h * 2 + 1];
        float sc2 = (q0 * k0 + q1 * k1) * scale;
        if (!sm.msk[j]) sc2 = -1.0e9f;
        float nm = fmaxf(mx, sc2);
        float corr = __expf(mx - nm);
        float e = __expf(sc2 - nm);
        ssum = ssum * corr + e;
        o0 = o0 * corr + e * sm.tmp[j * 32 + 12 + h * 2 + 0];
        o1 = o1 * corr + e * sm.tmp[j * 32 + 12 + h * 2 + 1];
        mx = nm;
      }
      float inv = 1.f / ssum;
      sm.obuf[i * 8 + h * 2 + 0] = o0 * inv;
      sm.obuf[i * 8 + h * 2 + 1] = o1 * inv;
    }
    __syncthreads();

    // out projection: (112x32pad) x (32x64pad)
    for (int idx = tid; idx < NPAD * 32; idx += BLOCK) {
      int r = idx >> 5, c = idx & 31;
      sm.ah[idx] = (r < NTOK && c < INNER) ? (_Float16)sm.obuf[r * 8 + c]
                                           : (_Float16)0.f;
    }
    stage_b_swz(sm.bh, out_w + l * INNER * EMBD, EMBD, 32, EMBD, INNER, EMBD, false);
    __syncthreads();
    lds_gemm(sm.ah, 32, sm.bh, sm.tmp, EMBD, NPAD, EMBD, 32);
    __syncthreads();
    for (int idx = tid; idx < NTOK * EMBD; idx += BLOCK)
      sm.xs[idx] += sm.tmp[idx] + out_b[l * EMBD + (idx % EMBD)];
    __syncthreads();

    // MLP: LN2 -> ff1 -> gelu -> ff2
    layernorm_to_f16(sm.xs, sm.ah, ln2_s + l * EMBD, ln2_b + l * EMBD);
    stage_b_swz(sm.bh, ff1_w + l * EMBD * MLPD, MLPD, EMBD, MLPD, EMBD, MLPD, false);
    __syncthreads();
    lds_gemm(sm.ah, EMBD, sm.bh, sm.tmp, MLPD, NPAD, MLPD, EMBD);
    __syncthreads();
    for (int idx = tid; idx < NPAD * 32; idx += BLOCK) {
      int r = idx >> 5, c = idx & 31;
      float g = 0.f;
      if (r < NTOK && c < MLPD) {
        float xg = sm.tmp[r * MLPD + c] + ff1_b[l * MLPD + c];
        g = 0.5f * xg * (1.f + erff(xg * 0.7071067811865476f));   // exact gelu
      }
      sm.ah[idx] = (_Float16)g;
    }
    stage_b_swz(sm.bh, ff2_w + l * MLPD * EMBD, EMBD, 32, EMBD, MLPD, EMBD, false);
    __syncthreads();
    lds_gemm(sm.ah, 32, sm.bh, sm.tmp, EMBD, NPAD, EMBD, 32);
    __syncthreads();
    for (int idx = tid; idx < NTOK * EMBD; idx += BLOCK)
      sm.xs[idx] += sm.tmp[idx] + ff2_b[l * EMBD + (idx % EMBD)];
    __syncthreads();
  }

  // ---- decoder: xd = relu(x @ dec_w + dec_b) -------------------------------
  for (int idx = tid; idx < NPAD * EMBD; idx += BLOCK) {
    int r = idx >> 6;
    sm.ah[idx] = (r < NTOK) ? (_Float16)sm.xs[idx] : (_Float16)0.f;
  }
  stage_b_swz(sm.bh, dec_w, OUTD, EMBD, OUTD, EMBD, OUTD, false);
  __syncthreads();
  lds_gemm(sm.ah, EMBD, sm.bh, sm.tmp, OUTD, NPAD, OUTD, EMBD);
  __syncthreads();
  for (int idx = tid; idx < NPAD * OUTD; idx += BLOCK) {
    int r = idx >> 5, c = idx & 31;
    float v2 = (r < NTOK) ? fmaxf(sm.tmp[idx] + dec_b[c], 0.f) : 0.f;
    sm.tmp[idx] = v2;                 // xd (fp32, kept for final contraction)
    sm.ah[idx]  = (_Float16)v2;       // A operand for K gemm
  }
  // B = ot_w^T (32x25 -> 32x32 padded): B[k][m] = ot_w[m][k]
  stage_b_swz(sm.bh, ot_w, OUTD, 32, 32, 32, OTM, true);
  __syncthreads();
  float* Kmat = sm.tmp + NPAD * OUTD;
  lds_gemm(sm.ah, OUTD, sm.bh, Kmat, 32, NPAD, 32, OUTD);          // K = xd @ ot_w^T
  __syncthreads();

  // ---- K filter + Sinkhorn (100 iterations, all in LDS) --------------------
  for (int idx = tid; idx < NTOK * OTM; idx += BLOCK) {
    int r = idx / OTM, m = idx % OTM;
    float pa = (float)(r + 1) / (float)NTOK;
    float pb = (float)(m + 1) / (float)OTM;
    float f = (fabsf(pa - pb) < 0.1f) ? 1.f : 0.f;
    Kmat[r * 32 + m] = Kmat[r * 32 + m] * f * 10.f;                // * filt / eps
  }
  for (int r = tid; r < NPAD; r += BLOCK) sm.u[r] = 0.f;
  if (tid < 32) sm.vv[tid] = 0.f;
  __syncthreads();

  const float logb = -3.2188758248682006f;                          // -log(25)
  for (int it = 0; it < OTI; ++it) {
    // u update: 100 rows, 25-wide logsumexp each
    for (int r = tid; r < NTOK; r += BLOCK) {
      float mx = -3.0e38f;
#pragma unroll
      for (int m = 0; m < OTM; ++m) mx = fmaxf(mx, Kmat[r * 32 + m] + sm.vv[m]);
      float ssum = 0.f;
#pragma unroll
      for (int m = 0; m < OTM; ++m) ssum += __expf(Kmat[r * 32 + m] + sm.vv[m] - mx);
      sm.u[r] = sm.loga[r] - (mx + __logf(ssum));
    }
    __syncthreads();
    // v update: 25 cols x 100 rows, 4-way split reduction (100 active lanes)
    if (tid < 100) {
      int m = tid % OTM, part = tid / OTM;
      int r0 = part * 25;
      float mx = -3.0e38f;
#pragma unroll
      for (int r = 0; r < 25; ++r)
        mx = fmaxf(mx, Kmat[(r0 + r) * 32 + m] + sm.u[r0 + r]);
      float ssum = 0.f;
#pragma unroll
      for (int r = 0; r < 25; ++r)
        ssum += __expf(Kmat[(r0 + r) * 32 + m] + sm.u[r0 + r] - mx);
      sm.pmx[part * 32 + m]  = mx;
      sm.psum[part * 32 + m] = ssum;
    }
    __syncthreads();
    if (tid < OTM) {
      float m0 = sm.pmx[tid],      m1 = sm.pmx[32 + tid];
      float m2 = sm.pmx[64 + tid], m3 = sm.pmx[96 + tid];
      float M = fmaxf(fmaxf(m0, m1), fmaxf(m2, m3));
      float S = sm.psum[tid]      * __expf(m0 - M)
              + sm.psum[32 + tid] * __expf(m1 - M)
              + sm.psum[64 + tid] * __expf(m2 - M)
              + sm.psum[96 + tid] * __expf(m3 - M);
      sm.vv[tid] = logb - (M + __logf(S));
    }
    __syncthreads();
  }

  // ---- out = (T*25)^T @ xd : (32x128pad) x (128x32) via WMMA ---------------
  for (int idx = tid; idx < 32 * 128; idx += BLOCK) {
    int m = idx >> 7, nn = idx & 127;
    float tv = 0.f;
    if (m < OTM && nn < NTOK)
      tv = 25.f * __expf(Kmat[nn * 32 + m] + sm.u[nn] + sm.vv[m]);
    sm.ah[idx] = (_Float16)tv;
  }
  stage_b_swz(sm.bh, sm.tmp, 32, 128, 32, NTOK, 32, false);         // B = xd
  __syncthreads();
  lds_gemm(sm.ah, 128, sm.bh, sm.xs, 32, 32, 32, 128);
  __syncthreads();
  for (int i2 = tid; i2 < OTM * OUTD; i2 += BLOCK)
    out[s * (OTM * OUTD) + i2] = sm.xs[i2];
}

extern "C" void kernel_launch(void* const* d_in, const int* in_sizes, int n_in,
                              void* d_out, int out_size, void* d_ws, size_t ws_size,
                              hipStream_t stream) {
  (void)in_sizes; (void)n_in; (void)out_size; (void)d_ws; (void)ws_size;
  seq_forward<<<800, BLOCK, 0, stream>>>(
      (const float*)d_in[0],  (const int*)d_in[1],    (const float*)d_in[2],
      (const float*)d_in[3],  (const float*)d_in[4],  (const float*)d_in[5],
      (const float*)d_in[6],  (const float*)d_in[7],  (const float*)d_in[8],
      (const float*)d_in[9],  (const float*)d_in[10], (const float*)d_in[11],
      (const float*)d_in[12], (const float*)d_in[13], (const float*)d_in[14],
      (const float*)d_in[15], (const float*)d_in[16], (const float*)d_in[17],
      (const float*)d_in[18], (const float*)d_in[19], (float*)d_out);
}
